// DecoderBlock_12773232738395
// MI455X (gfx1250) — matrix operand order repaired and checked by reading, not compiled, not run on hardware
//
#include <hip/hip_runtime.h>
#include <hip/hip_bf16.h>
#include <math.h>

// ---------------------------------------------------------------------------
// CDNA5 (gfx1250) WMMA types
// ---------------------------------------------------------------------------
typedef __attribute__((ext_vector_type(16))) __bf16 v16bf;
typedef __attribute__((ext_vector_type(8)))  float  v8f;

#define NQ    900
#define CHN   256
#define EMBD  512
#define SENC  2500
#define HEADS 8
#define LD_SA 912    // 900 keys padded to mult. of 16
#define LD_CA 2512   // 2500 keys padded to mult. of 16
#define LAMW  0.5f
#define EPSLN 1e-5f

// ---------------------------------------------------------------------------
// Generic batched GEMM:  C = alpha * (A @ B) [+ bias] [relu]
//   A: M x K   row-major, lda (mult of 4), per-batch stride sA (elements)
//   B: transB ? (N x K, ldb) : (K x N, ldb), per-batch stride sB
//   C: M x N   row-major, ldc, per-batch stride sC
//
// Wave tile 32x32 (2x2 WMMA frags, 4 x v_wmma_f32_16x16x32_bf16 per K-chunk);
// block = 2x2 waves = 64x64 tile.  K loop split: guard-free fast path over
// K & ~31 (float4 loads), one guarded remainder chunk.  Out-of-range rows /
// cols are clamped for address safety only -- their results are never stored.
// ---------------------------------------------------------------------------
__global__ void k_gemm(const float* __restrict__ A, int lda, long sA,
                       const float* __restrict__ B, int ldb, long sB, int transB,
                       const float* __restrict__ bias,
                       float* __restrict__ C, int ldc, long sC,
                       int M, int N, int K, float alpha, int relu)
{
    const int z = blockIdx.z;
    A += (long)z * sA;
    B += (long)z * sB;
    C += (long)z * sC;

    const int lane = threadIdx.x & 31;
    const int wv   = threadIdx.x >> 5;
    const int m0   = blockIdx.y * 64 + (wv >> 1) * 32;
    const int n0   = blockIdx.x * 64 + (wv & 1) * 32;

    // 16-bit A (16x32), contiguous k order used here: lanes 0-15 row M=lane, K = akb..akb+15
    // with akb=0; lanes 16-31 same rows with akb=16.
    const int arow0 = m0 + (lane & 15);
    const int arow1 = arow0 + 16;
    const int akb   = (lane >> 4) * 16;
    // 16-bit B (32x16), same contiguous k order as A: lanes 0-15 col N=lane, K=0..15;
    // lanes 16-31 col N=lane-16, K=16..31.
    const int bcol0 = n0 + (lane & 15);
    const int bcol1 = bcol0 + 16;
    const int bkb   = (lane >> 4) * 16;

    // address-safe clamped indices (garbage rows/cols are never stored)
    const size_t ra0 = (size_t)(arow0 < M ? arow0 : M - 1) * lda;
    const size_t ra1 = (size_t)(arow1 < M ? arow1 : M - 1) * lda;
    const int    bc0 = bcol0 < N ? bcol0 : N - 1;
    const int    bc1 = bcol1 < N ? bcol1 : N - 1;

    v8f acc00 = {}, acc01 = {}, acc10 = {}, acc11 = {};

    const int Kfull = K & ~31;

    for (int k0 = 0; k0 < Kfull; k0 += 32) {
        if (k0 + 32 < K)  // speculative prefetch -> global_prefetch_b8
            __builtin_prefetch(A + ra0 + k0 + 32 + akb, 0, 1);

        v16bf a0, a1, b0, b1;
        // ---- A fragments: two contiguous 8-float runs per lane ----
        {
            const float4* p = (const float4*)(A + ra0 + k0 + akb);
            float4 x0 = p[0], x1 = p[1];
            const float4* q = (const float4*)(A + ra0 + k0 + akb + 8);
            float4 y0 = q[0], y1 = q[1];
            a0[0]=(__bf16)x0.x; a0[1]=(__bf16)x0.y; a0[2]=(__bf16)x0.z; a0[3]=(__bf16)x0.w;
            a0[4]=(__bf16)x1.x; a0[5]=(__bf16)x1.y; a0[6]=(__bf16)x1.z; a0[7]=(__bf16)x1.w;
            a0[8]=(__bf16)y0.x; a0[9]=(__bf16)y0.y; a0[10]=(__bf16)y0.z; a0[11]=(__bf16)y0.w;
            a0[12]=(__bf16)y1.x; a0[13]=(__bf16)y1.y; a0[14]=(__bf16)y1.z; a0[15]=(__bf16)y1.w;
        }
        {
            const float4* p = (const float4*)(A + ra1 + k0 + akb);
            float4 x0 = p[0], x1 = p[1];
            const float4* q = (const float4*)(A + ra1 + k0 + akb + 8);
            float4 y0 = q[0], y1 = q[1];
            a1[0]=(__bf16)x0.x; a1[1]=(__bf16)x0.y; a1[2]=(__bf16)x0.z; a1[3]=(__bf16)x0.w;
            a1[4]=(__bf16)x1.x; a1[5]=(__bf16)x1.y; a1[6]=(__bf16)x1.z; a1[7]=(__bf16)x1.w;
            a1[8]=(__bf16)y0.x; a1[9]=(__bf16)y0.y; a1[10]=(__bf16)y0.z; a1[11]=(__bf16)y0.w;
            a1[12]=(__bf16)y1.x; a1[13]=(__bf16)y1.y; a1[14]=(__bf16)y1.z; a1[15]=(__bf16)y1.w;
        }
        // ---- B fragments ----
        if (transB) {
            const float4* p = (const float4*)(B + (size_t)bc0 * ldb + k0 + bkb);
            float4 x0 = p[0], x1 = p[1], x2 = p[2], x3 = p[3];
            b0[0]=(__bf16)x0.x; b0[1]=(__bf16)x0.y; b0[2]=(__bf16)x0.z; b0[3]=(__bf16)x0.w;
            b0[4]=(__bf16)x1.x; b0[5]=(__bf16)x1.y; b0[6]=(__bf16)x1.z; b0[7]=(__bf16)x1.w;
            b0[8]=(__bf16)x2.x; b0[9]=(__bf16)x2.y; b0[10]=(__bf16)x2.z; b0[11]=(__bf16)x2.w;
            b0[12]=(__bf16)x3.x; b0[13]=(__bf16)x3.y; b0[14]=(__bf16)x3.z; b0[15]=(__bf16)x3.w;
            const float4* q = (const float4*)(B + (size_t)bc1 * ldb + k0 + bkb);
            float4 y0 = q[0], y1 = q[1], y2 = q[2], y3 = q[3];
            b1[0]=(__bf16)y0.x; b1[1]=(__bf16)y0.y; b1[2]=(__bf16)y0.z; b1[3]=(__bf16)y0.w;
            b1[4]=(__bf16)y1.x; b1[5]=(__bf16)y1.y; b1[6]=(__bf16)y1.z; b1[7]=(__bf16)y1.w;
            b1[8]=(__bf16)y2.x; b1[9]=(__bf16)y2.y; b1[10]=(__bf16)y2.z; b1[11]=(__bf16)y2.w;
            b1[12]=(__bf16)y3.x; b1[13]=(__bf16)y3.y; b1[14]=(__bf16)y3.z; b1[15]=(__bf16)y3.w;
        } else {
#pragma unroll
            for (int i = 0; i < 16; ++i) {
                size_t rk = (size_t)(k0 + bkb + i) * ldb;
                b0[i] = (__bf16)B[rk + bc0];
                b1[i] = (__bf16)B[rk + bc1];
            }
        }
        acc00 = __builtin_amdgcn_wmma_f32_16x16x32_bf16(false, a0, false, b0, (short)0, acc00, false, false);
        acc01 = __builtin_amdgcn_wmma_f32_16x16x32_bf16(false, a0, false, b1, (short)0, acc01, false, false);
        acc10 = __builtin_amdgcn_wmma_f32_16x16x32_bf16(false, a1, false, b0, (short)0, acc10, false, false);
        acc11 = __builtin_amdgcn_wmma_f32_16x16x32_bf16(false, a1, false, b1, (short)0, acc11, false, false);
    }

    if (Kfull < K) {   // single guarded remainder chunk
        const int k0 = Kfull;
        v16bf a0, a1, b0, b1;
#pragma unroll
        for (int i = 0; i < 8; ++i) {
            int kk  = k0 + akb + i;
            int kk2 = kk + 8;
            a0[i]     = (__bf16)((kk  < K) ? A[ra0 + kk ] : 0.0f);
            a0[i + 8] = (__bf16)((kk2 < K) ? A[ra0 + kk2] : 0.0f);
            a1[i]     = (__bf16)((kk  < K) ? A[ra1 + kk ] : 0.0f);
            a1[i + 8] = (__bf16)((kk2 < K) ? A[ra1 + kk2] : 0.0f);
        }
#pragma unroll
        for (int i = 0; i < 16; ++i) {
            int kk = k0 + bkb + i;
            float x0 = 0.0f, x1 = 0.0f;
            if (kk < K) {
                if (transB) { x0 = B[(size_t)bc0 * ldb + kk]; x1 = B[(size_t)bc1 * ldb + kk]; }
                else        { x0 = B[(size_t)kk * ldb + bc0]; x1 = B[(size_t)kk * ldb + bc1]; }
            }
            b0[i] = (__bf16)x0;
            b1[i] = (__bf16)x1;
        }
        acc00 = __builtin_amdgcn_wmma_f32_16x16x32_bf16(false, a0, false, b0, (short)0, acc00, false, false);
        acc01 = __builtin_amdgcn_wmma_f32_16x16x32_bf16(false, a0, false, b1, (short)0, acc01, false, false);
        acc10 = __builtin_amdgcn_wmma_f32_16x16x32_bf16(false, a1, false, b0, (short)0, acc10, false, false);
        acc11 = __builtin_amdgcn_wmma_f32_16x16x32_bf16(false, a1, false, b1, (short)0, acc11, false, false);
    }

    // C/D f32 16x16 layout: lane 0-15 -> N=lane, M=r; lanes 16-31 -> M=r+8.
    const int cnl   = lane & 15;
    const int rbase = (lane >> 4) * 8;
#pragma unroll
    for (int t = 0; t < 4; ++t) {
        const v8f& acc = (t == 0) ? acc00 : (t == 1) ? acc01 : (t == 2) ? acc10 : acc11;
        int cn = n0 + ((t & 1) * 16) + cnl;
        int cmb = m0 + ((t >> 1) * 16) + rbase;
        if (cn < N) {
            float bv = bias ? bias[cn] : 0.0f;
#pragma unroll
            for (int r = 0; r < 8; ++r) {
                int cm = cmb + r;
                if (cm < M) {
                    float vv = acc[r] * alpha + bv;
                    if (relu) vv = fmaxf(vv, 0.0f);
                    C[(size_t)cm * ldc + cn] = vv;
                }
            }
        }
    }
}

// ---------------------------------------------------------------------------
// Row softmax, one wave (32 lanes) per row.  mode==1 subtracts pairwise
// euclidean distance bias computed from obj_coords.  Pads cols [nk,ld) = 0.
// grid = (rows, heads), block = 32
// ---------------------------------------------------------------------------
__global__ void k_softmax(const float* __restrict__ S, long headStride,
                          float* __restrict__ P,
                          const float* __restrict__ coords, int mode,
                          int nk, int ld)
{
    const int row  = blockIdx.x;
    const int h    = blockIdx.y;
    const int lane = threadIdx.x;
    const float* s = S + (long)h * headStride + (long)row * ld;
    float*       p = P + (long)h * headStride + (long)row * ld;

    float qx = 0.0f, qy = 0.0f;
    if (mode) { qx = coords[row * 2 + 0]; qy = coords[row * 2 + 1]; }

    float mx = -3.0e38f;
    for (int k = lane; k < nk; k += 32) {
        float v = s[k];
        if (mode) {
            float dx = qx - coords[k * 2 + 0];
            float dy = qy - coords[k * 2 + 1];
            v -= sqrtf(dx * dx + dy * dy + 1e-12f);
        }
        mx = fmaxf(mx, v);
    }
#pragma unroll
    for (int off = 16; off; off >>= 1) mx = fmaxf(mx, __shfl_xor(mx, off, 32));

    float sum = 0.0f;
    for (int k = lane; k < nk; k += 32) {
        float v = s[k];
        if (mode) {
            float dx = qx - coords[k * 2 + 0];
            float dy = qy - coords[k * 2 + 1];
            v -= sqrtf(dx * dx + dy * dy + 1e-12f);
        }
        sum += __expf(v - mx);
    }
#pragma unroll
    for (int off = 16; off; off >>= 1) sum += __shfl_xor(sum, off, 32);
    float inv = 1.0f / sum;

    for (int k = lane; k < ld; k += 32) {
        if (k < nk) {
            float v = s[k];
            if (mode) {
                float dx = qx - coords[k * 2 + 0];
                float dy = qy - coords[k * 2 + 1];
                v -= sqrtf(dx * dx + dy * dy + 1e-12f);
            }
            p[k] = __expf(v - mx) * inv;
        } else {
            p[k] = 0.0f;
        }
    }
}

// ---------------------------------------------------------------------------
// out = LAM * LN(base + a1; g1,b1) + (1-LAM) * LN(base + a2; g2,b2), row=512
// grid = (rows), block = 32
// ---------------------------------------------------------------------------
__global__ void k_ln_mix(const float* __restrict__ base,
                         const float* __restrict__ a1,
                         const float* __restrict__ a2,
                         const float* __restrict__ g1, const float* __restrict__ b1,
                         const float* __restrict__ g2, const float* __restrict__ b2,
                         float* __restrict__ out, int cols)
{
    const int row  = blockIdx.x;
    const int lane = threadIdx.x;
    const size_t roff = (size_t)row * cols;

    float s1 = 0, s2 = 0, q1 = 0, q2 = 0;
    for (int c = lane; c < cols; c += 32) {
        float x1 = base[roff + c] + a1[roff + c];
        float x2 = base[roff + c] + a2[roff + c];
        s1 += x1; q1 += x1 * x1;
        s2 += x2; q2 += x2 * x2;
    }
#pragma unroll
    for (int off = 16; off; off >>= 1) {
        s1 += __shfl_xor(s1, off, 32); q1 += __shfl_xor(q1, off, 32);
        s2 += __shfl_xor(s2, off, 32); q2 += __shfl_xor(q2, off, 32);
    }
    float inv_n = 1.0f / (float)cols;
    float m1 = s1 * inv_n, m2 = s2 * inv_n;
    float v1 = q1 * inv_n - m1 * m1;
    float v2 = q2 * inv_n - m2 * m2;
    float i1 = rsqrtf(v1 + EPSLN);
    float i2 = rsqrtf(v2 + EPSLN);

    for (int c = lane; c < cols; c += 32) {
        float x1 = base[roff + c] + a1[roff + c];
        float x2 = base[roff + c] + a2[roff + c];
        float y1 = (x1 - m1) * i1 * g1[c] + b1[c];
        float y2 = (x2 - m2) * i2 * g2[c] + b2[c];
        out[roff + c] = LAMW * y1 + (1.0f - LAMW) * y2;
    }
}

// ---------------------------------------------------------------------------
// out = LN(A + Badd; g,b).  Strided in/out so we can slice columns of `o`
// and write straight into the concatenated d_out.   grid=(rows), block=32
// ---------------------------------------------------------------------------
__global__ void k_ln(const float* __restrict__ A, int lda,
                     const float* __restrict__ Badd, int ldb,
                     const float* __restrict__ g, const float* __restrict__ bb,
                     float* __restrict__ out, int ldo, int cols)
{
    const int row  = blockIdx.x;
    const int lane = threadIdx.x;

    float s = 0, q = 0;
    for (int c = lane; c < cols; c += 32) {
        float x = A[(size_t)row * lda + c] + Badd[(size_t)row * ldb + c];
        s += x; q += x * x;
    }
#pragma unroll
    for (int off = 16; off; off >>= 1) {
        s += __shfl_xor(s, off, 32); q += __shfl_xor(q, off, 32);
    }
    float inv_n = 1.0f / (float)cols;
    float m = s * inv_n;
    float v = q * inv_n - m * m;
    float iv = rsqrtf(v + EPSLN);

    for (int c = lane; c < cols; c += 32) {
        float x = A[(size_t)row * lda + c] + Badd[(size_t)row * ldb + c];
        out[(size_t)row * ldo + c] = (x - m) * iv * g[c] + bb[c];
    }
}

// X[r, c] += P[r, c % 256]   (adds concat([q_pos, q_pos]) to a 900x512 matrix)
__global__ void k_add_dup(float* __restrict__ X, const float* __restrict__ P,
                          int rows)
{
    int idx = blockIdx.x * 256 + threadIdx.x;
    if (idx < rows * EMBD) {
        int r = idx >> 9, c = idx & 511;
        X[idx] += P[(size_t)r * CHN + (c & 255)];
    }
}

// out[r, h*64+j] = j<32 ? X[r, h*32+j] : Y[r, h*32+j-32]  (per-head concat)
__global__ void k_interleave(const float* __restrict__ X, int ldx,
                             const float* __restrict__ Y, int ldy,
                             float* __restrict__ out, int rows)
{
    int idx = blockIdx.x * 256 + threadIdx.x;
    if (idx < rows * EMBD) {
        int r = idx >> 9, c = idx & 511;
        int h = c >> 6, j = c & 63;
        float v = (j < 32) ? X[(size_t)r * ldx + h * 32 + j]
                           : Y[(size_t)r * ldy + h * 32 + (j - 32)];
        out[idx] = v;
    }
}

// ---------------------------------------------------------------------------
// Host-side orchestration
// ---------------------------------------------------------------------------
static inline void gemm(hipStream_t st,
                        const float* A, int lda, long sA,
                        const float* B, int ldb, long sB, int transB,
                        const float* bias,
                        float* C, int ldc, long sC,
                        int M, int N, int K, float alpha, int relu, int batch)
{
    dim3 grid((N + 63) / 64, (M + 63) / 64, batch);
    k_gemm<<<grid, 128, 0, st>>>(A, lda, sA, B, ldb, sB, transB,
                                 bias, C, ldc, sC, M, N, K, alpha, relu);
}

extern "C" void kernel_launch(void* const* d_in, const int* in_sizes, int n_in,
                              void* d_out, int out_size, void* d_ws, size_t ws_size,
                              hipStream_t stream)
{
    (void)in_sizes; (void)n_in; (void)out_size; (void)ws_size;

    // input order mirrors setup_inputs() flattening
    const float* object_queries = (const float*)d_in[0];   // 900 x 512
    const float* encoder_output = (const float*)d_in[1];   // 2500 x 256
    const float* obj_coords     = (const float*)d_in[2];   // 900 x 2
    const float* obj_pos_embed  = (const float*)d_in[3];   // 900 x 256
    const float* obj_sin_embed  = (const float*)d_in[4];   // 900 x 256
    const float* pe2d           = (const float*)d_in[5];   // 2500 x 256
    // sa
    const float* sa_Wq_obj = (const float*)d_in[6];
    const float* sa_Wq_pos = (const float*)d_in[7];
    const float* sa_Wk_obj = (const float*)d_in[8];
    const float* sa_Wk_pos = (const float*)d_in[9];
    const float* sa_Wv_obj = (const float*)d_in[10];
    const float* sa_Wo_sa  = (const float*)d_in[11];
    const float* sa_bo_sa  = (const float*)d_in[12];
    const float* sa_Wo_pa  = (const float*)d_in[13];
    const float* sa_bo_pa  = (const float*)d_in[14];
    const float* sa_ln1_g  = (const float*)d_in[15];
    const float* sa_ln1_b  = (const float*)d_in[16];
    const float* sa_ln2_g  = (const float*)d_in[17];
    const float* sa_ln2_b  = (const float*)d_in[18];
    // ca
    const float* ca_Wq_obj = (const float*)d_in[19];
    const float* ca_Wq_pos = (const float*)d_in[20];
    const float* ca_Wk_enc = (const float*)d_in[21];
    const float* ca_Wk_pos = (const float*)d_in[22];
    const float* ca_Wv_enc = (const float*)d_in[23];
    // branch params: cls at 24..33, reg at 34..43
    // order: Wo, bo, W1, b1, W2, b2, n1_g, n1_b, n2_g, n2_b

    float* out = (float*)d_out;                             // 900 x 512

    // ---------------- workspace carve-out (floats) ----------------
    float* w = (float*)d_ws;
    auto carve = [&](size_t n) { float* p = w; w += n; return p; };
    float* qpos  = carve((size_t)NQ * CHN);
    float* kpos  = carve((size_t)NQ * CHN);
    float* q     = carve((size_t)NQ * EMBD);
    float* k     = carve((size_t)NQ * EMBD);
    float* v     = carve((size_t)NQ * EMBD);
    float* attn  = carve((size_t)NQ * EMBD);      // reused for both SA variants
    float* o1    = carve((size_t)NQ * EMBD);
    float* o2    = carve((size_t)NQ * EMBD);
    float* o     = carve((size_t)NQ * EMBD);
    float* qobj  = carve((size_t)NQ * EMBD);
    float* qpos2 = carve((size_t)NQ * CHN);
    float* qcls  = carve((size_t)NQ * EMBD);
    float* qreg  = carve((size_t)NQ * EMBD);
    float* kenc  = carve((size_t)SENC * CHN);
    float* kpos2 = carve((size_t)SENC * CHN);
    float* kall  = carve((size_t)SENC * EMBD);
    float* v2    = carve((size_t)SENC * CHN);
    float* sco   = carve((size_t)HEADS * NQ * LD_SA);   // also reused for CA scores
    float* prb   = carve((size_t)HEADS * NQ * LD_SA);   // also reused for CA probs
    float* caout = carve((size_t)NQ * CHN);
    float* ca2   = carve((size_t)NQ * CHN);
    float* xbuf  = carve((size_t)NQ * CHN);
    float* h1    = carve((size_t)NQ * 4 * CHN);
    float* rbuf  = carve((size_t)NQ * CHN);

    const long hsSA = (long)NQ * LD_SA;
    const int eltBlocks = (NQ * EMBD + 255) / 256;

    // ================= self attention =================
    gemm(stream, obj_pos_embed, CHN, 0, sa_Wq_pos, CHN, 0, 0, nullptr,
         qpos, CHN, 0, NQ, CHN, CHN, 1.0f, 0, 1);
    gemm(stream, obj_pos_embed, CHN, 0, sa_Wk_pos, CHN, 0, 0, nullptr,
         kpos, CHN, 0, NQ, CHN, CHN, 1.0f, 0, 1);

    gemm(stream, object_queries, EMBD, 0, sa_Wq_obj, EMBD, 0, 0, nullptr,
         q, EMBD, 0, NQ, EMBD, EMBD, 1.0f, 0, 1);
    k_add_dup<<<eltBlocks, 256, 0, stream>>>(q, qpos, NQ);
    gemm(stream, object_queries, EMBD, 0, sa_Wk_obj, EMBD, 0, 0, nullptr,
         k, EMBD, 0, NQ, EMBD, EMBD, 1.0f, 0, 1);
    k_add_dup<<<eltBlocks, 256, 0, stream>>>(k, kpos, NQ);
    gemm(stream, object_queries, EMBD, 0, sa_Wv_obj, EMBD, 0, 0, nullptr,
         v, EMBD, 0, NQ, EMBD, EMBD, 1.0f, 0, 1);

    // scores[h] = Q_h @ K_h^T / 8   (batched over 8 heads, head dim 64)
    gemm(stream, q, EMBD, 64, k, EMBD, 64, 1, nullptr,
         sco, LD_SA, hsSA, NQ, NQ, 64, 0.125f, 0, HEADS);

    // ---- variant 1: plain softmax ----
    { dim3 g(NQ, HEADS);
      k_softmax<<<g, 32, 0, stream>>>(sco, hsSA, prb, nullptr, 0, NQ, LD_SA); }
    gemm(stream, prb, LD_SA, hsSA, v, EMBD, 64, 0, nullptr,
         attn, EMBD, 64, NQ, 64, NQ, 1.0f, 0, HEADS);
    gemm(stream, attn, EMBD, 0, sa_Wo_sa, EMBD, 0, 0, sa_bo_sa,
         o1, EMBD, 0, NQ, EMBD, EMBD, 1.0f, 0, 1);

    // ---- variant 2: distance-biased softmax ----
    { dim3 g(NQ, HEADS);
      k_softmax<<<g, 32, 0, stream>>>(sco, hsSA, prb, obj_coords, 1, NQ, LD_SA); }
    gemm(stream, prb, LD_SA, hsSA, v, EMBD, 64, 0, nullptr,
         attn, EMBD, 64, NQ, 64, NQ, 1.0f, 0, HEADS);
    gemm(stream, attn, EMBD, 0, sa_Wo_pa, EMBD, 0, 0, sa_bo_pa,
         o2, EMBD, 0, NQ, EMBD, EMBD, 1.0f, 0, 1);

    // o = LAM*LN(x+o1) + (1-LAM)*LN(x+o2)
    k_ln_mix<<<NQ, 32, 0, stream>>>(object_queries, o1, o2,
                                    sa_ln1_g, sa_ln1_b, sa_ln2_g, sa_ln2_b,
                                    o, EMBD);

    // ================= cross attention setup =================
    gemm(stream, o, EMBD, 0, ca_Wq_obj, EMBD, 0, 0, nullptr,
         qobj, EMBD, 0, NQ, EMBD, EMBD, 1.0f, 0, 1);
    gemm(stream, obj_sin_embed, CHN, 0, ca_Wq_pos, CHN, 0, 0, nullptr,
         qpos2, CHN, 0, NQ, CHN, CHN, 1.0f, 0, 1);
    k_interleave<<<eltBlocks, 256, 0, stream>>>(qobj, EMBD, qpos2, CHN, qcls, NQ);
    k_interleave<<<eltBlocks, 256, 0, stream>>>(qobj + CHN, EMBD, qpos2, CHN, qreg, NQ);

    gemm(stream, encoder_output, CHN, 0, ca_Wk_enc, CHN, 0, 0, nullptr,
         kenc, CHN, 0, SENC, CHN, CHN, 1.0f, 0, 1);
    gemm(stream, pe2d, CHN, 0, ca_Wk_pos, CHN, 0, 0, nullptr,
         kpos2, CHN, 0, SENC, CHN, CHN, 1.0f, 0, 1);
    { int nb = (SENC * EMBD + 255) / 256;
      k_interleave<<<nb, 256, 0, stream>>>(kenc, CHN, kpos2, CHN, kall, SENC); }
    gemm(stream, encoder_output, CHN, 0, ca_Wv_enc, CHN, 0, 0, nullptr,
         v2, CHN, 0, SENC, CHN, CHN, 1.0f, 0, 1);

    const float ca_scale = 1.0f / sqrtf((float)EMBD);

    // ================= two branches =================
    for (int br = 0; br < 2; ++br) {
        const float* q_     = br ? qreg : qcls;
        int base            = br ? 34 : 24;
        const float* Wo  = (const float*)d_in[base + 0];
        const float* bo  = (const float*)d_in[base + 1];
        const float* W1  = (const float*)d_in[base + 2];
        const float* b1  = (const float*)d_in[base + 3];
        const float* W2  = (const float*)d_in[base + 4];
        const float* b2  = (const float*)d_in[base + 5];
        const float* n1g = (const float*)d_in[base + 6];
        const float* n1b = (const float*)d_in[base + 7];
        const float* n2g = (const float*)d_in[base + 8];
        const float* n2b = (const float*)d_in[base + 9];
        const float* inp = o + (br ? CHN : 0);     // o[..., :256] / o[..., 256:]
        float* outCol    = out + (br ? CHN : 0);

        // sc = q_ @ k_all^T / sqrt(512)
        gemm(stream, q_, EMBD, 0, kall, EMBD, 0, 1, nullptr,
             sco, LD_CA, 0, NQ, SENC, EMBD, ca_scale, 0, 1);
        { dim3 g(NQ, 1);
          k_softmax<<<g, 32, 0, stream>>>(sco, 0, prb, nullptr, 0, SENC, LD_CA); }
        // P @ V2
        gemm(stream, prb, LD_CA, 0, v2, CHN, 0, 0, nullptr,
             caout, CHN, 0, NQ, CHN, SENC, 1.0f, 0, 1);
        // output projection
        gemm(stream, caout, CHN, 0, Wo, CHN, 0, 0, bo,
             ca2, CHN, 0, NQ, CHN, CHN, 1.0f, 0, 1);
        // x = LN(inputs + ca_out)
        k_ln<<<NQ, 32, 0, stream>>>(inp, EMBD, ca2, CHN, n1g, n1b,
                                    xbuf, CHN, CHN);
        // FFN
        gemm(stream, xbuf, CHN, 0, W1, 4 * CHN, 0, 0, b1,
             h1, 4 * CHN, 0, NQ, 4 * CHN, CHN, 1.0f, 1, 1);
        gemm(stream, h1, 4 * CHN, 0, W2, CHN, 0, 0, b2,
             rbuf, CHN, 0, NQ, CHN, 4 * CHN, 1.0f, 0, 1);
        // out slice = LN(x + r)
        k_ln<<<NQ, 32, 0, stream>>>(xbuf, CHN, rbuf, CHN, n2g, n2b,
                                    outCol, EMBD, CHN);
    }
}
